// SoftPoolFeat_72112500900172
// MI455X (gfx1250) — matrix operand order repaired and checked by reading, compiled, not run on hardware
//
#include <hip/hip_runtime.h>
#include <cstdint>
#include <cstddef>

// ---------------------------------------------------------------------------
// SoftPoolFeat forward for MI455X (gfx1250, wave32, WMMA).
// GEMMs: LDS-staged bf16 tiles -> v_wmma_f32_16x16x32_bf16 (f32 accumulate).
// All K/O are compile-time so the hot loop has zero bounds checks; no bias in
// GEMM (the only biased GEMM's bias is folded into the argmax consumer).
// ---------------------------------------------------------------------------

typedef __bf16 bf16_t;
typedef __attribute__((ext_vector_type(16))) __bf16 v16bf;
typedef __attribute__((ext_vector_type(8)))  __bf16 v8bf;
typedef __attribute__((ext_vector_type(8)))  float  v8f;

#define B_   4
#define N_   8192
#define R_   16

__device__ __forceinline__ unsigned short bfbits(float f) {
  unsigned u = __builtin_bit_cast(unsigned, f);
  return (unsigned short)((u + 0x7FFFu + ((u >> 16) & 1u)) >> 16);   // RNE
}
__device__ __forceinline__ unsigned pack2(float a, float b) {
  return (unsigned)bfbits(a) | ((unsigned)bfbits(b) << 16);
}

// ---------------------------------------------------------------------------
// WMMA GEMM:  Z[b,o,n] = sum_k W[o,k] * X[b,k,n]
// Block = 128 threads = 4 waves. Block tile: 64 rows x 64 cols
// (wave w owns rows mb+16w..mb+16w+15; all waves share the B tile).
// grid = (N/64, ceil(O/64), B).  K % 32 == 0 (inputs pre-padded).
// ---------------------------------------------------------------------------
template <int K, int O>
__global__ __launch_bounds__(128) void k_gemm_wmma(
    const float* __restrict__ W, const float* __restrict__ X, float* __restrict__ Z)
{
  __shared__ v8bf aA[64 * 4];   // 64 rows x 32 k  (bf16, [row][k])
  __shared__ v8bf bB[64 * 4];   // 64 cols x 32 k  (bf16, [n][k])
  bf16_t* aAe = reinterpret_cast<bf16_t*>(aA);
  bf16_t* bBe = reinterpret_cast<bf16_t*>(bB);

  const int t    = threadIdx.x;
  const int lane = t & 31;
  const int wave = t >> 5;
  const int hi   = lane >> 4;
  const int lo   = lane & 15;
  const int nb   = blockIdx.x << 6;
  const int mb   = blockIdx.y << 6;
  const float* Xb = X + (size_t)blockIdx.z * K * N_;
  float*       Zb = Z + (size_t)blockIdx.z * O * N_;

  v8f acc0 = {}, acc1 = {}, acc2 = {}, acc3 = {};

  for (int k0 = 0; k0 < K; k0 += 32) {
    // ---- stage A tile: 64x32 f32 -> bf16 (coalesced float4 along k) ----
#pragma unroll
    for (int it = 0; it < 4; ++it) {
      int row = it * 16 + (t >> 3);        // 0..63
      int kc  = (t & 7) * 4;               // 0,4,..,28
      float4 w4 = make_float4(0.f, 0.f, 0.f, 0.f);
      if ((O % 64 == 0) || (mb + row) < O)
        w4 = *(const float4*)(W + (size_t)(mb + row) * K + k0 + kc);
      uint2 u; u.x = pack2(w4.x, w4.y); u.y = pack2(w4.z, w4.w);
      *(uint2*)(aAe + row * 32 + kc) = u;
    }
    // ---- stage B tile: 32x64 f32 -> bf16, [n][k] layout ----
#pragma unroll
    for (int it = 0; it < 4; ++it) {
      int task = t + it * 128;             // 0..511
      int n    = task & 63;
      int kg   = task >> 6;                // 0..7 (4 k's each)
      const float* src = Xb + (size_t)(k0 + kg * 4) * N_ + nb + n;
      float x0 = src[0];
      float x1 = src[N_];
      float x2 = src[2 * N_];
      float x3 = src[3 * N_];
      uint2 u; u.x = pack2(x0, x1); u.y = pack2(x2, x3);
      *(uint2*)(bBe + n * 32 + kg * 4) = u;
    }
    __syncthreads();

    // ---- fragments from LDS (two b128 each) ----
    int row = (wave << 4) + lo;
    v8bf aL = aA[row * 4 + hi * 2];
    v8bf aH = aA[row * 4 + hi * 2 + 1];
    v16bf a = __builtin_shufflevector(aL, aH, 0,1,2,3,4,5,6,7,8,9,10,11,12,13,14,15);

    v8bf p0 = bB[(0 * 16 + lo) * 4 + hi * 2], q0 = bB[(0 * 16 + lo) * 4 + hi * 2 + 1];
    v8bf p1 = bB[(1 * 16 + lo) * 4 + hi * 2], q1 = bB[(1 * 16 + lo) * 4 + hi * 2 + 1];
    v8bf p2 = bB[(2 * 16 + lo) * 4 + hi * 2], q2 = bB[(2 * 16 + lo) * 4 + hi * 2 + 1];
    v8bf p3 = bB[(3 * 16 + lo) * 4 + hi * 2], q3 = bB[(3 * 16 + lo) * 4 + hi * 2 + 1];
    v16bf b0 = __builtin_shufflevector(p0, q0, 0,1,2,3,4,5,6,7,8,9,10,11,12,13,14,15);
    v16bf b1 = __builtin_shufflevector(p1, q1, 0,1,2,3,4,5,6,7,8,9,10,11,12,13,14,15);
    v16bf b2 = __builtin_shufflevector(p2, q2, 0,1,2,3,4,5,6,7,8,9,10,11,12,13,14,15);
    v16bf b3 = __builtin_shufflevector(p3, q3, 0,1,2,3,4,5,6,7,8,9,10,11,12,13,14,15);

    acc0 = __builtin_amdgcn_wmma_f32_16x16x32_bf16(false, a, false, b0, (short)0, acc0, false, false);
    acc1 = __builtin_amdgcn_wmma_f32_16x16x32_bf16(false, a, false, b1, (short)0, acc1, false, false);
    acc2 = __builtin_amdgcn_wmma_f32_16x16x32_bf16(false, a, false, b2, (short)0, acc2, false, false);
    acc3 = __builtin_amdgcn_wmma_f32_16x16x32_bf16(false, a, false, b3, (short)0, acc3, false, false);
    __syncthreads();
  }

  // C/D layout: VGPR r -> (lanes 0-15: M=r) (lanes 16-31: M=8+r), N = lane&15
#pragma unroll
  for (int r = 0; r < 8; ++r) {
    int orow = mb + (wave << 4) + hi * 8 + r;
    if ((O % 64 == 0) || orow < O) {
      float* dst = Zb + (size_t)orow * N_ + nb + lo;
      dst[0]  = acc0[r];
      dst[16] = acc1[r];
      dst[32] = acc2[r];
      dst[48] = acc3[r];
    }
  }
}

// ---------------------------------------------------------------------------
// zero-pad helpers (K=3 / K=19 layers are padded to K=32 so GEMMs stay clean)
// ---------------------------------------------------------------------------
__global__ void k_pad_w(const float* __restrict__ Win, float* __restrict__ Wout, int O, int Kin, int Kout)
{
  int i = blockIdx.x * blockDim.x + threadIdx.x;
  if (i >= O * Kout) return;
  int o = i / Kout, k = i % Kout;
  Wout[i] = (k < Kin) ? Win[o * Kin + k] : 0.f;
}

__global__ void k_pad_x(const float* __restrict__ in, float* __restrict__ out, int Cin, int Cout)
{
  size_t i = (size_t)blockIdx.x * blockDim.x + threadIdx.x;
  size_t total = (size_t)B_ * Cout * N_;
  if (i >= total) return;
  int n = (int)(i % N_);
  int c = (int)((i / N_) % Cout);
  int b = (int)(i / ((size_t)Cout * N_));
  out[i] = (c < Cin) ? in[((size_t)b * Cin + c) * N_ + n] : 0.f;
}

// ---------------------------------------------------------------------------
// BN batch stats over (B, N) per channel:  mean[c], rstd[c]
// ---------------------------------------------------------------------------
__global__ __launch_bounds__(256) void k_bn_stats(
    const float* __restrict__ z, float* __restrict__ mean, float* __restrict__ rstd, int C)
{
  __shared__ float ss[256], sq[256];
  int c = blockIdx.x;
  float s = 0.f, q = 0.f;
  for (int b = 0; b < B_; ++b) {
    const float* p = z + ((size_t)b * C + c) * N_;
    for (int n = threadIdx.x; n < N_; n += 256) { float v = p[n]; s += v; q += v * v; }
  }
  ss[threadIdx.x] = s; sq[threadIdx.x] = q;
  __syncthreads();
  for (int st = 128; st > 0; st >>= 1) {
    if (threadIdx.x < st) { ss[threadIdx.x] += ss[threadIdx.x + st]; sq[threadIdx.x] += sq[threadIdx.x + st]; }
    __syncthreads();
  }
  if (threadIdx.x == 0) {
    float cnt = (float)(B_ * N_);
    float m   = ss[0] / cnt;
    float var = sq[0] / cnt - m * m;
    mean[c] = m;
    rstd[c] = rsqrtf(var + 1e-5f);
  }
}

__global__ void k_bn_apply(const float* __restrict__ z, float* __restrict__ out,
                           const float* __restrict__ mean, const float* __restrict__ rstd,
                           const float* __restrict__ g, const float* __restrict__ be,
                           int C, int relu)
{
  size_t i = (size_t)blockIdx.x * blockDim.x + threadIdx.x;
  size_t total = (size_t)B_ * C * N_;
  if (i >= total) return;
  int c = (int)((i / N_) % C);
  float v = (z[i] - mean[c]) * rstd[c] * g[c] + be[c];
  if (relu) v = fmaxf(v, 0.f);
  out[i] = v;
}

// Fused BN + ReLU + max over N (STN s3 channel chunks)
__global__ __launch_bounds__(256) void k_bnrelu_max(
    const float* __restrict__ z, const float* __restrict__ mean, const float* __restrict__ rstd,
    const float* __restrict__ g, const float* __restrict__ be,
    float* __restrict__ hmax, int C, int chanOff, int Ctot)
{
  __shared__ float sm[256];
  int b = blockIdx.x / C;
  int c = blockIdx.x % C;
  const float* p = z + ((size_t)b * C + c) * N_;
  float a = rstd[c] * g[c];
  float m = 0.f;                       // post-ReLU values are >= 0
  for (int n = threadIdx.x; n < N_; n += 256) {
    float v = (p[n] - mean[c]) * a + be[c];
    m = fmaxf(m, v);
  }
  sm[threadIdx.x] = m;
  __syncthreads();
  for (int st = 128; st > 0; st >>= 1) {
    if (threadIdx.x < st) sm[threadIdx.x] = fmaxf(sm[threadIdx.x], sm[threadIdx.x + st]);
    __syncthreads();
  }
  if (threadIdx.x == 0) hmax[(size_t)b * Ctot + chanOff + c] = sm[0];
}

// ---------------------------------------------------------------------------
// Bitonic argsort (descending, stable) of 8192 keys fully in LDS (64 KB).
// One workgroup per (b, r).
// ---------------------------------------------------------------------------
__global__ __launch_bounds__(1024) void k_sort(const float* __restrict__ val, int* __restrict__ order)
{
  __shared__ float key[N_];
  __shared__ int   idx[N_];
  int b = blockIdx.x >> 4;
  int r = blockIdx.x & 15;
  const float* v = val + ((size_t)b * R_ + r) * N_;
  for (int i = threadIdx.x; i < N_; i += 1024) { key[i] = v[i]; idx[i] = i; }
  __syncthreads();
  for (int k = 2; k <= N_; k <<= 1) {
    for (int j = k >> 1; j > 0; j >>= 1) {
      for (int t = threadIdx.x; t < (N_ >> 1); t += 1024) {
        int i = ((t & ~(j - 1)) << 1) | (t & (j - 1));
        int l = i | j;
        bool up = ((i & k) == 0);
        float ki = key[i], kl = key[l];
        int   ii = idx[i], il = idx[l];
        bool aFirst = (ki > kl) || (ki == kl && ii < il);
        bool doswap = up ? !aFirst : aFirst;
        if (doswap) { key[i] = kl; key[l] = ki; idx[i] = il; idx[l] = ii; }
      }
      __syncthreads();
    }
  }
  int* o = order + ((size_t)b * R_ + r) * N_;
  for (int i = threadIdx.x; i < N_; i += 1024) o[i] = idx[i];
}

// sp_idx[b, j, r, n] = float(order[b, r, n])
__global__ void k_sp_idx(const int* __restrict__ order, float* __restrict__ out)
{
  size_t i = (size_t)blockIdx.x * blockDim.x + threadIdx.x;
  const size_t total = (size_t)B_ * 19 * R_ * N_;
  if (i >= total) return;
  int n = (int)(i & (N_ - 1));
  int r = (int)((i >> 13) & 15);
  int t = (int)(i >> 17);              // t = b*19 + j
  int b = t / 19;
  out[i] = (float)order[(((size_t)b * R_ + r) << 13) + n];
}

// pws = concat(one_hot(argmax_r (val+bsort)), x) : (B, 19, N)
// bsort is applied here (per-region constant shift does not affect the sort).
__global__ __launch_bounds__(256) void k_build_pws(
    const float* __restrict__ val, const float* __restrict__ bsort,
    const float* __restrict__ x, float* __restrict__ pws)
{
  int g = blockIdx.x;
  int b = g >> 5;
  int n = ((g & 31) << 8) + threadIdx.x;
  const float* vb = val + (size_t)b * R_ * N_ + n;
  float best = vb[0] + bsort[0]; int bi = 0;
#pragma unroll
  for (int r = 1; r < R_; ++r) {
    float vv = vb[(size_t)r * N_] + bsort[r];
    if (vv > best) { best = vv; bi = r; }
  }
  float* pb = pws + (size_t)b * 19 * N_ + n;
#pragma unroll
  for (int r = 0; r < R_; ++r) pb[(size_t)r * N_] = (r == bi) ? 1.f : 0.f;
  const float* xb = x + (size_t)b * 3 * N_ + n;
#pragma unroll
  for (int c = 0; c < 3; ++c) pb[(size_t)(16 + c) * N_] = xb[(size_t)c * N_];
}

// featT[b, n, c] = feat[b, c, n]
__global__ void k_transpose(const float* __restrict__ feat, float* __restrict__ featT)
{
  size_t i = (size_t)blockIdx.x * blockDim.x + threadIdx.x;
  const size_t total = (size_t)B_ * 256 * N_;
  if (i >= total) return;
  int c = (int)(i & 255);
  int n = (int)((i >> 8) & (N_ - 1));
  int b = (int)(i >> 21);
  featT[i] = feat[(((size_t)b * 256 + c) << 13) + n];
}

// cabins[b,c,r,k] = max_{j<1024} feat[b, c, order[b,r,k*1024+j]]
__global__ __launch_bounds__(256) void k_cabins(
    const float* __restrict__ featT, const int* __restrict__ order, float* __restrict__ cabins)
{
  __shared__ int sidx[1024];
  int blk = blockIdx.x;
  int kk = blk & 7;
  int r  = (blk >> 3) & 15;
  int b  = blk >> 7;
  const int* o = order + (((size_t)b * R_ + r) << 13) + kk * 1024;
  for (int j = threadIdx.x; j < 1024; j += 256) sidx[j] = o[j];
  __syncthreads();
  int c = threadIdx.x;
  float m = -3.4e38f;
  for (int j = 0; j < 1024; ++j) {
    float v = featT[(((size_t)b << 13) + sidx[j]) * 256 + c];
    m = fmaxf(m, v);
  }
  cabins[(((size_t)b * 256 + c) * R_ + r) * 8 + kk] = m;
}

// 1xTW conv along width (compile-time Win/TW so inner loops unroll fully)
template <int Win, int TW>
__global__ void k_conv_row(const float* __restrict__ in, const float* __restrict__ W,
                           const float* __restrict__ bias, float* __restrict__ out)
{
  constexpr int Wout = Win - TW + 1;
  int i = blockIdx.x * blockDim.x + threadIdx.x;
  int total = B_ * 256 * R_ * Wout;
  if (i >= total) return;
  int w = i % Wout;
  int r = (i / Wout) & 15;
  int o = (i / (Wout * R_)) & 255;
  int b = i / (Wout * R_ * 256);
  float acc = bias[o];
  const float* ib = in + ((size_t)b * 256 * R_ + r) * Win + w;
  const float* wo = W + (size_t)o * 256 * TW;
  for (int c = 0; c < 256; ++c) {
    const float* ic = ib + (size_t)c * R_ * Win;
    const float* wc = wo + (size_t)c * TW;
#pragma unroll
    for (int t = 0; t < TW; ++t) acc += ic[t] * wc[t];
  }
  out[(((size_t)b * 256 + o) * R_ + r) * Wout + w] = acc;
}

// station[b,o] = bc5[o] + sum_{c,r} trains[b,c,r] * Wc5[o,c,r]
__global__ void k_station(const float* __restrict__ trains, const float* __restrict__ W,
                          const float* __restrict__ bias, float* __restrict__ out)
{
  int i = blockIdx.x * blockDim.x + threadIdx.x;
  if (i >= B_ * 256) return;
  int b = i >> 8, o = i & 255;
  float acc = bias[o];
  const float* tb = trains + (size_t)b * 256 * R_;
  const float* wo = W + (size_t)o * 256 * R_;
  for (int k = 0; k < 256 * R_; ++k) acc += tb[k] * wo[k];
  out[i] = acc;
}

// Tiny FC: out[b,o] = bias[o] + sum_k in[b,k] * W[o,k]
__global__ void k_fc(const float* __restrict__ in, const float* __restrict__ W,
                     const float* __restrict__ bias, float* __restrict__ out, int O, int K)
{
  int i = blockIdx.x * blockDim.x + threadIdx.x;
  if (i >= B_ * O) return;
  int b = i / O, o = i % O;
  float acc = bias[o];
  const float* xr = in + (size_t)b * K;
  const float* wr = W + (size_t)o * K;
  for (int k = 0; k < K; ++k) acc += xr[k] * wr[k];
  out[i] = acc;
}

// BN over batch dim only (B = 4), in place, optional ReLU
__global__ void k_bn0_apply(float* __restrict__ h, const float* __restrict__ g,
                            const float* __restrict__ be, int O, int relu)
{
  int o = blockIdx.x * blockDim.x + threadIdx.x;
  if (o >= O) return;
  float m = 0.f;
  for (int b = 0; b < B_; ++b) m += h[(size_t)b * O + o];
  m *= 0.25f;
  float v = 0.f;
  for (int b = 0; b < B_; ++b) { float d = h[(size_t)b * O + o] - m; v += d * d; }
  v *= 0.25f;
  float rs = rsqrtf(v + 1e-5f);
  for (int b = 0; b < B_; ++b) {
    float val = (h[(size_t)b * O + o] - m) * rs * g[o] + be[o];
    if (relu) val = fmaxf(val, 0.f);
    h[(size_t)b * O + o] = val;
  }
}

// trans[b, i, j] = fc3[b, i*19+j] + I(i==j)
__global__ void k_trans_out(const float* __restrict__ fc3, float* __restrict__ out)
{
  int i = blockIdx.x * blockDim.x + threadIdx.x;
  if (i >= B_ * 361) return;
  int e = i % 361;
  out[i] = fc3[i] + ((e % 20 == 0) ? 1.f : 0.f);
}

// feature (B, 1043, 16, 256) assembly
__global__ void k_feature(const float* __restrict__ featT, const int* __restrict__ order,
                          const float* __restrict__ cabins, const float* __restrict__ trains,
                          const float* __restrict__ station, const float* __restrict__ pws,
                          float* __restrict__ out)
{
  size_t i = (size_t)blockIdx.x * blockDim.x + threadIdx.x;
  const size_t total = (size_t)B_ * 1043 * R_ * 256;
  if (i >= total) return;
  int s  = (int)(i & 255);
  int r  = (int)((i >> 8) & 15);
  int t  = (int)(i >> 12);
  int cc = t % 1043;
  int b  = t / 1043;
  float v;
  if (cc < 256) {
    int idx = order[(((size_t)b * R_ + r) << 13) + s];
    v = featT[(((size_t)b << 13) + idx) * 256 + cc];
  } else if (cc < 512) {
    v = cabins[(((size_t)b * 256 + (cc - 256)) * R_ + r) * 8 + 0];
  } else if (cc < 768) {
    v = trains[((size_t)b * 256 + (cc - 512)) * R_ + r];
  } else if (cc < 1024) {
    v = station[(size_t)b * 256 + (cc - 768)];
  } else {
    int idx = order[(((size_t)b * R_ + r) << 13) + s];
    v = pws[((size_t)b * 19 + (cc - 1024)) * N_ + idx];
  }
  out[i] = v;
}

// ---------------------------------------------------------------------------
extern "C" void kernel_launch(void* const* d_in, const int* in_sizes, int n_in,
                              void* d_out, int out_size, void* d_ws, size_t ws_size,
                              hipStream_t stream)
{
  (void)in_sizes; (void)n_in; (void)out_size; (void)ws_size;

  const float* x     = (const float*)d_in[0];
  const float* W1    = (const float*)d_in[1];
  const float* ga1   = (const float*)d_in[3];
  const float* be1   = (const float*)d_in[4];
  const float* W2    = (const float*)d_in[5];
  const float* ga2   = (const float*)d_in[7];
  const float* be2   = (const float*)d_in[8];
  const float* W3    = (const float*)d_in[9];
  const float* ga3   = (const float*)d_in[11];
  const float* be3   = (const float*)d_in[12];
  const float* Wsort = (const float*)d_in[13];
  const float* bsort = (const float*)d_in[14];
  const float* Wc1   = (const float*)d_in[15];
  const float* bc1   = (const float*)d_in[16];
  const float* Wc2   = (const float*)d_in[17];
  const float* bc2   = (const float*)d_in[18];
  const float* Wc3   = (const float*)d_in[19];
  const float* bc3   = (const float*)d_in[20];
  const float* Wc5   = (const float*)d_in[21];
  const float* bc5   = (const float*)d_in[22];
  const float* sW1   = (const float*)d_in[23];
  const float* sg1   = (const float*)d_in[25];
  const float* sbe1  = (const float*)d_in[26];
  const float* sW2   = (const float*)d_in[27];
  const float* sg2   = (const float*)d_in[29];
  const float* sbe2  = (const float*)d_in[30];
  const float* sW3   = (const float*)d_in[31];
  const float* sg3   = (const float*)d_in[33];
  const float* sbe3  = (const float*)d_in[34];
  const float* fW1   = (const float*)d_in[35];
  const float* fb1   = (const float*)d_in[36];
  const float* fg1   = (const float*)d_in[37];
  const float* fbe1  = (const float*)d_in[38];
  const float* fW2   = (const float*)d_in[39];
  const float* fb2   = (const float*)d_in[40];
  const float* fg2   = (const float*)d_in[41];
  const float* fbe2  = (const float*)d_in[42];
  const float* fW3   = (const float*)d_in[43];
  const float* fb3   = (const float*)d_in[44];

  float* out          = (float*)d_out;
  float* out_feature  = out;                   // (4,1043,16,256)
  float* out_cabins   = out + 17088512;        // (4,256,16,8)
  float* out_spidx    = out + 17219584;        // (4,19,16,8192)
  float* out_trans    = out + 27181056;        // (4,19,19)

  float* ws = (float*)d_ws;
  size_t off = 0;
  auto alloc = [&](size_t n) { float* p = ws + off; off += (n + 63) & ~(size_t)63; return p; };
  float* bufA    = alloc((size_t)B_ * 256 * N_);
  float* bufB    = alloc((size_t)B_ * 128 * N_);
  float* feat    = alloc((size_t)B_ * 256 * N_);
  float* featT   = alloc((size_t)B_ * 256 * N_);
  float* xp      = alloc((size_t)B_ * 32 * N_);    // x padded 3 -> 32
  float* pwsp    = alloc((size_t)B_ * 32 * N_);    // pws padded 19 -> 32
  float* val     = alloc((size_t)B_ * R_ * N_);
  int*   order   = (int*)alloc((size_t)B_ * R_ * N_);
  float* pws     = alloc((size_t)B_ * 19 * N_);
  float* trains1 = alloc((size_t)B_ * 256 * R_ * 6);
  float* trains2 = alloc((size_t)B_ * 256 * R_ * 4);
  float* trains  = alloc((size_t)B_ * 256 * R_);
  float* station = alloc((size_t)B_ * 256);
  float* hmax    = alloc((size_t)B_ * 1024);
  float* fc1b    = alloc((size_t)B_ * 512);
  float* fc2b    = alloc((size_t)B_ * 256);
  float* fc3b    = alloc((size_t)B_ * 361);
  float* W1p     = alloc(64 * 32);
  float* sW1p    = alloc(64 * 32);
  float* m1 = alloc(64);   float* r1 = alloc(64);
  float* m2 = alloc(128);  float* r2 = alloc(128);
  float* m3 = alloc(256);  float* r3 = alloc(256);
  float* ms1 = alloc(64);  float* rs1 = alloc(64);
  float* ms2 = alloc(128); float* rs2 = alloc(128);
  float* ms3 = alloc(1024);float* rs3 = alloc(1024);

  const int GX = N_ / 64;                       // 128 col-blocks

  // ---- pconv1: 3 -> 64 (padded to K=32), BN+ReLU ----
  k_pad_w<<<(64 * 32 + 255) / 256, 256, 0, stream>>>(W1, W1p, 64, 3, 32);
  {
    size_t tot = (size_t)B_ * 32 * N_;
    k_pad_x<<<(unsigned)((tot + 255) / 256), 256, 0, stream>>>(x, xp, 3, 32);
  }
  k_gemm_wmma<32, 64><<<dim3(GX, 1, B_), 128, 0, stream>>>(W1p, xp, bufA);
  k_bn_stats<<<64, 256, 0, stream>>>(bufA, m1, r1, 64);
  {
    size_t tot = (size_t)B_ * 64 * N_;
    k_bn_apply<<<(unsigned)((tot + 255) / 256), 256, 0, stream>>>(bufA, bufB, m1, r1, ga1, be1, 64, 1);
  }
  // ---- pconv2: 64 -> 128 ----
  k_gemm_wmma<64, 128><<<dim3(GX, 2, B_), 128, 0, stream>>>(W2, bufB, bufA);
  k_bn_stats<<<128, 256, 0, stream>>>(bufA, m2, r2, 128);
  {
    size_t tot = (size_t)B_ * 128 * N_;
    k_bn_apply<<<(unsigned)((tot + 255) / 256), 256, 0, stream>>>(bufA, bufB, m2, r2, ga2, be2, 128, 1);
  }
  // ---- pconv3: 128 -> 256, BN (no ReLU) -> feat ----
  k_gemm_wmma<128, 256><<<dim3(GX, 4, B_), 128, 0, stream>>>(W3, bufB, bufA);
  k_bn_stats<<<256, 256, 0, stream>>>(bufA, m3, r3, 256);
  {
    size_t tot = (size_t)B_ * 256 * N_;
    k_bn_apply<<<(unsigned)((tot + 255) / 256), 256, 0, stream>>>(bufA, feat, m3, r3, ga3, be3, 256, 0);
  }
  // ---- region scores: val = Wsort @ feat (bsort folded into argmax) ----
  k_gemm_wmma<256, 16><<<dim3(GX, 1, B_), 128, 0, stream>>>(Wsort, feat, val);

  // ---- per-(b,r) descending argsort, sp_idx, pws ----
  k_sort<<<B_ * R_, 1024, 0, stream>>>(val, order);
  {
    size_t tot = (size_t)B_ * 19 * R_ * N_;
    k_sp_idx<<<(unsigned)((tot + 255) / 256), 256, 0, stream>>>(order, out_spidx);
  }
  k_build_pws<<<B_ * N_ / 256, 256, 0, stream>>>(val, bsort, x, pws);

  // ---- transpose feat, gather-max cabins ----
  {
    size_t tot = (size_t)B_ * 256 * N_;
    k_transpose<<<(unsigned)((tot + 255) / 256), 256, 0, stream>>>(feat, featT);
  }
  k_cabins<<<B_ * R_ * 8, 256, 0, stream>>>(featT, order, out_cabins);

  // ---- conv chain: cabins -> trains -> station ----
  k_conv_row<8, 3><<<(B_ * 256 * R_ * 6 + 255) / 256, 256, 0, stream>>>(out_cabins, Wc1, bc1, trains1);
  k_conv_row<6, 3><<<(B_ * 256 * R_ * 4 + 255) / 256, 256, 0, stream>>>(trains1, Wc2, bc2, trains2);
  k_conv_row<4, 4><<<(B_ * 256 * R_ * 1 + 255) / 256, 256, 0, stream>>>(trains2, Wc3, bc3, trains);
  k_station<<<(B_ * 256 + 255) / 256, 256, 0, stream>>>(trains, Wc5, bc5, station);

  // ---- STN: 19 -> 64 -> 128 -> 1024 (chunked) -> max ----
  k_pad_w<<<(64 * 32 + 255) / 256, 256, 0, stream>>>(sW1, sW1p, 64, 19, 32);
  {
    size_t tot = (size_t)B_ * 32 * N_;
    k_pad_x<<<(unsigned)((tot + 255) / 256), 256, 0, stream>>>(pws, pwsp, 19, 32);
  }
  k_gemm_wmma<32, 64><<<dim3(GX, 1, B_), 128, 0, stream>>>(sW1p, pwsp, bufA);
  k_bn_stats<<<64, 256, 0, stream>>>(bufA, ms1, rs1, 64);
  {
    size_t tot = (size_t)B_ * 64 * N_;
    k_bn_apply<<<(unsigned)((tot + 255) / 256), 256, 0, stream>>>(bufA, bufB, ms1, rs1, sg1, sbe1, 64, 1);
  }
  k_gemm_wmma<64, 128><<<dim3(GX, 2, B_), 128, 0, stream>>>(sW2, bufB, bufA);
  k_bn_stats<<<128, 256, 0, stream>>>(bufA, ms2, rs2, 128);
  {
    size_t tot = (size_t)B_ * 128 * N_;
    k_bn_apply<<<(unsigned)((tot + 255) / 256), 256, 0, stream>>>(bufA, bufB, ms2, rs2, sg2, sbe2, 128, 1);
  }
  for (int gi = 0; gi < 4; ++gi) {
    k_gemm_wmma<128, 256><<<dim3(GX, 4, B_), 128, 0, stream>>>(sW3 + (size_t)gi * 256 * 128, bufB, bufA);
    k_bn_stats<<<256, 256, 0, stream>>>(bufA, ms3 + gi * 256, rs3 + gi * 256, 256);
    k_bnrelu_max<<<B_ * 256, 256, 0, stream>>>(bufA, ms3 + gi * 256, rs3 + gi * 256,
                                               sg3 + gi * 256, sbe3 + gi * 256,
                                               hmax, 256, gi * 256, 1024);
  }
  // ---- STN FC head (tiny) ----
  k_fc<<<(B_ * 512 + 255) / 256, 256, 0, stream>>>(hmax, fW1, fb1, fc1b, 512, 1024);
  k_bn0_apply<<<(512 + 255) / 256, 256, 0, stream>>>(fc1b, fg1, fbe1, 512, 1);
  k_fc<<<(B_ * 256 + 255) / 256, 256, 0, stream>>>(fc1b, fW2, fb2, fc2b, 256, 512);
  k_bn0_apply<<<(256 + 255) / 256, 256, 0, stream>>>(fc2b, fg2, fbe2, 256, 1);
  k_fc<<<(B_ * 361 + 255) / 256, 256, 0, stream>>>(fc2b, fW3, fb3, fc3b, 361, 256);
  k_trans_out<<<(B_ * 361 + 255) / 256, 256, 0, stream>>>(fc3b, out_trans);

  // ---- final feature assembly ----
  {
    size_t tot = (size_t)B_ * 1043 * R_ * 256;
    k_feature<<<(unsigned)((tot + 255) / 256), 256, 0, stream>>>(featT, order, out_cabins, trains, station, pws, out_feature);
  }
}
